// TrustScorer_76836964926278
// MI455X (gfx1250) — matrix operand order repaired and checked by reading, not compiled, not run on hardware
//
#include <hip/hip_runtime.h>
#include <hip/hip_bf16.h>

#define DDIM 256
#define KSEQ 8

typedef __attribute__((ext_vector_type(16))) __bf16 v16bf;
typedef __attribute__((ext_vector_type(16))) unsigned short v16us;
typedef __attribute__((ext_vector_type(8)))  float v8f;

__device__ __forceinline__ unsigned short f2bf_bits(float f) {
  unsigned u = __builtin_bit_cast(unsigned int, f);
  return (unsigned short)((u + 0x7FFFu + ((u >> 16) & 1u)) >> 16);  // RNE
}
__device__ __forceinline__ float gelu_exact(float x) {
  return 0.5f * x * (1.0f + erff(x * 0.7071067811865475f));
}
__device__ __forceinline__ float sigmoidf(float x) {
  return 1.0f / (1.0f + __expf(-x));
}

// ---------------------------------------------------------------------------
// Repack Wq, Wk (256x256) and We1 (256x128) f32 row-major -> bf16 in WMMA
// B-tile order. Tile (ct,kt) = 512 bf16; element (lane*16+j):
//   col = lane%16, kk = (lane/16)*16 + j, src = W[(kt*32+kk)*Dout + ct*16+col]
// ---------------------------------------------------------------------------
__global__ void repack_kernel(const float* __restrict__ Wq,
                              const float* __restrict__ Wk,
                              const float* __restrict__ We1,
                              unsigned short* __restrict__ wsq,
                              unsigned short* __restrict__ wsk,
                              unsigned short* __restrict__ wse1) {
  int idx = blockIdx.x * blockDim.x + threadIdx.x;
  if (idx < 65536) {  // 16 col-tiles x 8 k-tiles x 512 (Dout=256)
    int t = idx >> 9, ct = t >> 3, kt = t & 7;
    int e = idx & 511, ln = e >> 4, j = e & 15;
    int col = ln & 15, kk = ((ln >> 4) << 4) + j;
    int src = (kt * 32 + kk) * 256 + ct * 16 + col;
    wsq[idx] = f2bf_bits(Wq[src]);
    wsk[idx] = f2bf_bits(Wk[src]);
  }
  if (idx < 32768) {  // 8 col-tiles x 8 k-tiles x 512 (Dout=128)
    int t = idx >> 9, ct = t >> 3, kt = t & 7;
    int e = idx & 511, ln = e >> 4, j = e & 15;
    int col = ln & 15, kk = ((ln >> 4) << 4) + j;
    int src = (kt * 32 + kk) * 128 + ct * 16 + col;
    wse1[idx] = f2bf_bits(We1[src]);
  }
}

// Build a 16x32 bf16 A tile: lane = row (m=lane&15), contiguous K split:
// lane<16: K = {kt*32+0..15}; lane>=16: K = {kt*32+16..31}
__device__ __forceinline__ v16bf load_a_tile(const float* __restrict__ row,
                                             int kt, int hi8) {
  v16us a;
#pragma unroll
  for (int g = 0; g < 2; ++g) {
    const float4* p = (const float4*)(row + kt * 32 + hi8 * 16 + g * 8);
    float4 f0 = p[0], f1 = p[1];
    a[g * 8 + 0] = f2bf_bits(f0.x); a[g * 8 + 1] = f2bf_bits(f0.y);
    a[g * 8 + 2] = f2bf_bits(f0.z); a[g * 8 + 3] = f2bf_bits(f0.w);
    a[g * 8 + 4] = f2bf_bits(f1.x); a[g * 8 + 5] = f2bf_bits(f1.y);
    a[g * 8 + 6] = f2bf_bits(f1.z); a[g * 8 + 7] = f2bf_bits(f1.w);
  }
  return __builtin_bit_cast(v16bf, a);
}

__device__ __forceinline__ v16bf load_b_tile(const unsigned short* __restrict__ w,
                                             int ctg, int kt, int lane) {
  const v16us* p = (const v16us*)(w + (((size_t)(ctg * 8 + kt)) << 9) + lane * 16);
  return __builtin_bit_cast(v16bf, *p);
}

// ---------------------------------------------------------------------------
// Activity score: one wave per edge pair. Q/K projections via WMMA bf16,
// per-head scores + softmax diagonal via VALU from LDS-staged head tiles.
// ---------------------------------------------------------------------------
__global__ __launch_bounds__(128, 1) void activity_kernel(
    const float* __restrict__ h_i, const float* __restrict__ h_j,
    const float* __restrict__ bq, const float* __restrict__ bk,
    const unsigned short* __restrict__ wsq, const unsigned short* __restrict__ wsk,
    float* __restrict__ ws_act, int nPairs, int nE) {
  __shared__ float ldsQ[4][16][68];
  __shared__ float ldsK[4][16][68];

  const int lane = threadIdx.x & 31;
  const int wv   = threadIdx.x >> 5;
  int pair = blockIdx.x * 4 + wv;
  if (pair >= nPairs) pair = nPairs - 1;
  const int e0  = pair * 2;
  const int m   = lane & 15;        // A-tile row
  const int hi8 = lane >> 4;
  const int edgeA = e0 + (m >> 3);  // rows 0-7 -> e0, 8-15 -> e0+1
  const int rowIn = m & 7;
  const float* rowI = h_i + ((size_t)edgeA * KSEQ + rowIn) * DDIM;
  const float* rowJ = h_j + ((size_t)edgeA * KSEQ + rowIn) * DDIM;

  v16bf Ai[8], Aj[8];
#pragma unroll
  for (int kt = 0; kt < 8; ++kt) {
    Ai[kt] = load_a_tile(rowI, kt, hi8);
    Aj[kt] = load_a_tile(rowJ, kt, hi8);
  }

  const int eq   = lane & 15;   // (edge,q) slot: 0-7 e0, 8-15 e1
  const int q    = eq & 7;
  const int eIdx = eq >> 3;
  const int half = lane >> 4;   // which 4 k's this lane computes
  float diagacc = 0.0f;

  for (int h = 0; h < 4; ++h) {
    // Q_h = h_i * Wq[:, h*64 : (h+1)*64] + bq  (4 col tiles x 8 k-steps)
#pragma unroll
    for (int ct = 0; ct < 4; ++ct) {
      const int colg = h * 4 + ct;
      const float bv = bq[colg * 16 + (lane & 15)];
      v8f acc;
#pragma unroll
      for (int r = 0; r < 8; ++r) acc[r] = bv;
#pragma unroll
      for (int kt = 0; kt < 8; ++kt) {
        v16bf b = load_b_tile(wsq, colg, kt, lane);
        acc = __builtin_amdgcn_wmma_f32_16x16x32_bf16(false, Ai[kt], false, b,
                                                      (short)0, acc, false, false);
      }
#pragma unroll
      for (int r = 0; r < 8; ++r)
        ldsQ[wv][r + hi8 * 8][ct * 16 + (lane & 15)] = acc[r];
    }
    // K_h = h_j * Wk[:, head cols] + bk
#pragma unroll
    for (int ct = 0; ct < 4; ++ct) {
      const int colg = h * 4 + ct;
      const float bv = bk[colg * 16 + (lane & 15)];
      v8f acc;
#pragma unroll
      for (int r = 0; r < 8; ++r) acc[r] = bv;
#pragma unroll
      for (int kt = 0; kt < 8; ++kt) {
        v16bf b = load_b_tile(wsk, colg, kt, lane);
        acc = __builtin_amdgcn_wmma_f32_16x16x32_bf16(false, Aj[kt], false, b,
                                                      (short)0, acc, false, false);
      }
#pragma unroll
      for (int r = 0; r < 8; ++r)
        ldsK[wv][r + hi8 * 8][ct * 16 + (lane & 15)] = acc[r];
    }
    // same-wave LDS visibility: wait all DS writes, block compiler reordering
    __asm volatile("s_wait_dscnt 0x0" ::: "memory");

    // scores s[e,h,q,k] = Q[q,:]·K[k,:] * HD^-0.5 ; lane covers 4 k's
    float s[4];
#pragma unroll
    for (int t = 0; t < 4; ++t) {
      const int kk = half * 4 + t;
      const float4* qr = (const float4*)&ldsQ[wv][eq][0];
      const float4* kr = (const float4*)&ldsK[wv][eIdx * 8 + kk][0];
      float d = 0.0f;
#pragma unroll
      for (int c = 0; c < 16; ++c) {
        float4 qa = qr[c], ka = kr[c];
        d += qa.x * ka.x + qa.y * ka.y + qa.z * ka.z + qa.w * ka.w;
      }
      s[t] = d * 0.125f;  // HD=64 -> scale 1/8
    }
    // softmax over 8 k's (split across lane and lane^16), take diagonal
    float mymax = fmaxf(fmaxf(s[0], s[1]), fmaxf(s[2], s[3]));
    float M = fmaxf(mymax, __shfl_xor(mymax, 16, 32));
    float se = __expf(s[0] - M) + __expf(s[1] - M) +
               __expf(s[2] - M) + __expf(s[3] - M);
    float tot = se + __shfl_xor(se, 16, 32);
    int qk = q - half * 4;
    if (qk >= 0 && qk < 4) diagacc += __expf(s[qk] - M) / tot;
  }

  // mean over 4 heads x 8 q = /32 ; reduce halves then q-groups of 8
  float v = diagacc + __shfl_xor(diagacc, 16, 32);
  v += __shfl_xor(v, 4, 32);
  v += __shfl_xor(v, 2, 32);
  v += __shfl_xor(v, 1, 32);
  if (lane == 0) ws_act[e0] = v * 0.03125f;
  if (lane == 8 && (e0 + 1) < nE) ws_act[e0 + 1] = v * 0.03125f;
}

// ---------------------------------------------------------------------------
// Edge plausibility MLP: 16 edges per wave, 256->128 via WMMA, exact GELU,
// dot with We2, sigmoid.
// ---------------------------------------------------------------------------
__global__ __launch_bounds__(128, 1) void edgemlp_kernel(
    const float* __restrict__ emb, const float* __restrict__ be1,
    const float* __restrict__ We2, const float* __restrict__ be2,
    const unsigned short* __restrict__ wse1, float* __restrict__ ws_edge,
    int nTiles, int nE) {
  const int lane = threadIdx.x & 31;
  const int wv   = threadIdx.x >> 5;
  int tile = blockIdx.x * 4 + wv;
  if (tile >= nTiles) tile = nTiles - 1;
  const int m = lane & 15;
  const int hi8 = lane >> 4;
  int edge = tile * 16 + m;
  if (edge >= nE) edge = nE - 1;
  const float* rowp = emb + (size_t)edge * DDIM;

  v16bf A[8];
#pragma unroll
  for (int kt = 0; kt < 8; ++kt) A[kt] = load_a_tile(rowp, kt, hi8);

  float part[8];
#pragma unroll
  for (int r = 0; r < 8; ++r) part[r] = 0.0f;

#pragma unroll
  for (int ct = 0; ct < 8; ++ct) {
    v8f acc;
#pragma unroll
    for (int r = 0; r < 8; ++r) acc[r] = 0.0f;
#pragma unroll
    for (int kt = 0; kt < 8; ++kt) {
      v16bf b = load_b_tile(wse1, ct, kt, lane);
      acc = __builtin_amdgcn_wmma_f32_16x16x32_bf16(false, A[kt], false, b,
                                                    (short)0, acc, false, false);
    }
    const int col = ct * 16 + (lane & 15);
    const float b1 = be1[col], w2 = We2[col];
#pragma unroll
    for (int r = 0; r < 8; ++r)
      part[r] += gelu_exact(acc[r] + b1) * w2;  // row r -> edge m=r+hi8*8
  }
  // reduce over the 16 column-lanes of each half-wave group
#pragma unroll
  for (int mask = 1; mask < 16; mask <<= 1)
#pragma unroll
    for (int r = 0; r < 8; ++r) part[r] += __shfl_xor(part[r], mask, 32);

  if ((lane & 15) == 0) {
    const float b2 = be2[0];
    const int rbase = tile * 16 + hi8 * 8;
#pragma unroll
    for (int r = 0; r < 8; ++r) {
      int e = rbase + r;
      if (e < nE) ws_edge[e] = sigmoidf(part[r] + b2);
    }
  }
}

// ---------------------------------------------------------------------------
// Cosine attr score + final 3->16->1 combiner. One warp per edge.
// ---------------------------------------------------------------------------
__global__ __launch_bounds__(256, 1) void final_kernel(
    const float* __restrict__ x_i, const float* __restrict__ x_j,
    const float* __restrict__ Wo1, const float* __restrict__ bo1,
    const float* __restrict__ Wo2, const float* __restrict__ bo2,
    const float* __restrict__ ws_act, const float* __restrict__ ws_edge,
    float* __restrict__ out, int nE) {
  const int lane = threadIdx.x & 31;
  const int wv   = threadIdx.x >> 5;
  int e = blockIdx.x * 8 + wv;
  if (e >= nE) e = nE - 1;
  const float4* xi = (const float4*)(x_i + (size_t)e * DDIM);
  const float4* xj = (const float4*)(x_j + (size_t)e * DDIM);
  float dp = 0.f, ni = 0.f, nj = 0.f;
#pragma unroll
  for (int p = 0; p < 2; ++p) {
    float4 a = xi[lane + p * 32], b = xj[lane + p * 32];
    dp += a.x * b.x + a.y * b.y + a.z * b.z + a.w * b.w;
    ni += a.x * a.x + a.y * a.y + a.z * a.z + a.w * a.w;
    nj += b.x * b.x + b.y * b.y + b.z * b.z + b.w * b.w;
  }
#pragma unroll
  for (int mask = 16; mask >= 1; mask >>= 1) {
    dp += __shfl_xor(dp, mask, 32);
    ni += __shfl_xor(ni, mask, 32);
    nj += __shfl_xor(nj, mask, 32);
  }
  float attr = (dp / (fmaxf(sqrtf(ni), 1e-8f) * fmaxf(sqrtf(nj), 1e-8f)) + 1.0f) * 0.5f;
  float s1 = ws_act[e], s2 = ws_edge[e];
  float o = bo2[0];
#pragma unroll
  for (int j = 0; j < 16; ++j) {
    float t = attr * Wo1[j] + s1 * Wo1[16 + j] + s2 * Wo1[32 + j] + bo1[j];
    o += gelu_exact(t) * Wo2[j];
  }
  if (lane == 0) out[e] = sigmoidf(o);
}

// ---------------------------------------------------------------------------
extern "C" void kernel_launch(void* const* d_in, const int* in_sizes, int n_in,
                              void* d_out, int out_size, void* d_ws, size_t ws_size,
                              hipStream_t stream) {
  (void)n_in; (void)out_size; (void)ws_size;
  const float* x_i  = (const float*)d_in[0];
  const float* x_j  = (const float*)d_in[1];
  const float* h_i  = (const float*)d_in[2];
  const float* h_j  = (const float*)d_in[3];
  const float* emb  = (const float*)d_in[4];
  const float* Wq   = (const float*)d_in[5];
  const float* bq   = (const float*)d_in[6];
  const float* Wk   = (const float*)d_in[7];
  const float* bk   = (const float*)d_in[8];
  const float* We1  = (const float*)d_in[9];
  const float* be1  = (const float*)d_in[10];
  const float* We2  = (const float*)d_in[11];
  const float* be2  = (const float*)d_in[12];
  const float* Wo1  = (const float*)d_in[13];
  const float* bo1  = (const float*)d_in[14];
  const float* Wo2  = (const float*)d_in[15];
  const float* bo2  = (const float*)d_in[16];
  const int nE = in_sizes[0] / DDIM;

  char* ws = (char*)d_ws;
  unsigned short* wsq  = (unsigned short*)(ws + 0);        // 256x256 bf16 = 128KB
  unsigned short* wsk  = (unsigned short*)(ws + 131072);   // 128KB
  unsigned short* wse1 = (unsigned short*)(ws + 262144);   // 256x128 bf16 = 64KB
  size_t actBytes = ((size_t)nE * 4 + 255) & ~(size_t)255;
  float* ws_act  = (float*)(ws + 327680);
  float* ws_edge = (float*)(ws + 327680 + actBytes);

  repack_kernel<<<256, 256, 0, stream>>>(Wq, Wk, We1, wsq, wsk, wse1);

  const int nPairs = (nE + 1) / 2;
  activity_kernel<<<(nPairs + 3) / 4, 128, 0, stream>>>(h_i, h_j, bq, bk, wsq, wsk,
                                                        ws_act, nPairs, nE);
  const int nTiles = (nE + 15) / 16;
  edgemlp_kernel<<<(nTiles + 3) / 4, 128, 0, stream>>>(emb, be1, We2, be2, wse1,
                                                       ws_edge, nTiles, nE);
  final_kernel<<<(nE + 7) / 8, 256, 0, stream>>>(x_i, x_j, Wo1, bo1, Wo2, bo2,
                                                 ws_act, ws_edge, (float*)d_out, nE);
}
